// APPNPGraphClassifier_45466523795734
// MI455X (gfx1250) — matrix operand order repaired and checked by reading, not compiled, not run on hardware
//
#include <hip/hip_runtime.h>

#define NN      100000
#define NE      1600000
#define NG      64
#define KSTEPS  10
#define ALPHA   0.1f
#define BNEPS   1e-5f

typedef __attribute__((ext_vector_type(16))) __bf16 v16bf;
typedef __attribute__((ext_vector_type(8)))  __bf16 v8bf;
typedef __attribute__((ext_vector_type(8)))  float  v8f;

// ---------------- fp32 -> bf16 conversion ----------------
__global__ void cvt_f32_bf16(const float* __restrict__ in, __bf16* __restrict__ out, int n) {
    int i = (blockIdx.x * blockDim.x + threadIdx.x) * 4;
    if (i + 3 < n) {
        float4 v = *(const float4*)(in + i);
        out[i + 0] = (__bf16)v.x;
        out[i + 1] = (__bf16)v.y;
        out[i + 2] = (__bf16)v.z;
        out[i + 3] = (__bf16)v.w;
    } else {
        for (; i < n; ++i) out[i] = (__bf16)in[i];
    }
}

// ------------- fused GEMM (bf16 WMMA) + bias + BN(eval) + ReLU -------------
// out[M,N] = relu(bn(A[M,K] @ W[N,K]^T + bias))
// One wave per 32-row tile (2 M-subtiles), computing the full 32 x (NT*16)
// panel. Each B fragment feeds two WMMAs -> 2x less B-feed bandwidth per FLOP.
template <int NT, bool OUT_BF16>
__global__ __launch_bounds__(256) void gemm_bn_relu(
    const __bf16* __restrict__ A, const __bf16* __restrict__ W,
    const float* __restrict__ bias,
    const float* __restrict__ g, const float* __restrict__ be,
    const float* __restrict__ rm, const float* __restrict__ rv,
    void* __restrict__ outp, int M, int K) {
    const int N    = NT * 16;
    const int lane = threadIdx.x & 31;
    const int tile = blockIdx.x * (blockDim.x >> 5) + (threadIdx.x >> 5);
    const int mtiles = M >> 5;            // 32 rows per wave
    if (tile >= mtiles) return;           // wave-uniform: EXEC stays all-1s for WMMA
    const int rowBase = tile * 32;

    // A-fragment: lane holds A[m][kc..kc+15] (contiguous k, matches B)
    const __bf16* arow0 = A + (size_t)(rowBase + (lane & 15)) * K + ((lane < 16) ? 0 : 16);
    const __bf16* arow1 = arow0 + (size_t)16 * K;
    // B-fragment: lane holds W[n][kc..kc+15]
    const __bf16* wbase = W + (size_t)(lane & 15) * K + ((lane < 16) ? 0 : 16);

    v8f acc0[NT] = {};
    v8f acc1[NT] = {};
    for (int k0 = 0; k0 < K; k0 += 32) {
        v8bf lo0 = *(const v8bf*)(arow0 + k0);
        v8bf hi0 = *(const v8bf*)(arow0 + k0 + 8);
        v8bf lo1 = *(const v8bf*)(arow1 + k0);
        v8bf hi1 = *(const v8bf*)(arow1 + k0 + 8);
        v16bf a0, a1;
#pragma unroll
        for (int i = 0; i < 8; ++i) {
            a0[i] = lo0[i]; a0[8 + i] = hi0[i];
            a1[i] = lo1[i]; a1[8 + i] = hi1[i];
        }
        const __bf16* wp = wbase + k0;
#pragma unroll
        for (int nt = 0; nt < NT; ++nt) {
            v16bf b = *(const v16bf*)(wp + (size_t)nt * 16 * K);
            acc0[nt] = __builtin_amdgcn_wmma_f32_16x16x32_bf16(
                false, a0, false, b, (short)0, acc0[nt], false, false);
            acc1[nt] = __builtin_amdgcn_wmma_f32_16x16x32_bf16(
                false, a1, false, b, (short)0, acc1[nt], false, false);
        }
    }

    // Epilogue: C/D layout -> row = rb + r + (lane>=16 ? 8:0), col = nt*16 + lane%16
    const int rowOff = (lane < 16) ? 0 : 8;
#pragma unroll
    for (int mt = 0; mt < 2; ++mt) {
        const v8f* accp = mt ? acc1 : acc0;
        const int rb = rowBase + mt * 16;
#pragma unroll
        for (int nt = 0; nt < NT; ++nt) {
            const int col = nt * 16 + (lane & 15);
            const float sc = g[col] * rsqrtf(rv[col] + BNEPS);
            const float sh = (bias[col] - rm[col]) * sc + be[col];
#pragma unroll
            for (int r = 0; r < 8; ++r) {
                float v = accp[nt][r] * sc + sh;
                v = fmaxf(v, 0.0f);
                const size_t idx = (size_t)(rb + r + rowOff) * N + col;
                if (OUT_BF16) ((__bf16*)outp)[idx] = (__bf16)v;
                else          ((float*)outp)[idx]  = v;
            }
        }
    }
}

// ---------------- APPNP helpers ----------------
__global__ void fill_f32(float* __restrict__ p, float v, int n) {
    int i = blockIdx.x * blockDim.x + threadIdx.x;
    if (i < n) p[i] = v;
}

__global__ void deg_edges(const int* __restrict__ col, float* __restrict__ deg, int nE) {
    int i = blockIdx.x * blockDim.x + threadIdx.x;
    if (i < nE) atomicAdd(deg + col[i], 1.0f);
}

__global__ void dinv_finalize(float* __restrict__ d, int n) {
    int i = blockIdx.x * blockDim.x + threadIdx.x;
    if (i < n) d[i] = rsqrtf(fmaxf(d[i], 1.0f));
}

// nxt = (1-a)*dinv[i]^2*cur + a*h   (self-loop term + teleport, writes every element)
__global__ void appnp_self(const float* __restrict__ cur, const float* __restrict__ h,
                           const float* __restrict__ dinv, float* __restrict__ nxt, int total) {
    int i = blockIdx.x * blockDim.x + threadIdx.x;
    if (i >= total) return;
    const float d = dinv[i >> 7];
    nxt[i] = (1.0f - ALPHA) * d * d * cur[i] + ALPHA * h[i];
}

// scatter-add over edges: nxt[col] += (1-a)*dinv[row]*dinv[col]*cur[row], 32 lanes/edge
__global__ __launch_bounds__(256) void appnp_edges(
    const int* __restrict__ row, const int* __restrict__ col,
    const float* __restrict__ dinv, const float* __restrict__ cur,
    float* __restrict__ nxt, int nE) {
    // edge index is wave-uniform: force into an SGPR so row/col become scalar loads
    const int e = __builtin_amdgcn_readfirstlane(
        (int)((blockIdx.x * blockDim.x + threadIdx.x) >> 5));
    const int lane = threadIdx.x & 31;
    if (e >= nE) return;
    const int r = row[e];
    const int c = col[e];
    const float w = (1.0f - ALPHA) * dinv[r] * dinv[c];
    const float4 v = *(const float4*)(cur + (size_t)r * 128 + lane * 4);
    float* dst = nxt + (size_t)c * 128 + lane * 4;
    atomicAdd(dst + 0, w * v.x);
    atomicAdd(dst + 1, w * v.y);
    atomicAdd(dst + 2, w * v.z);
    atomicAdd(dst + 3, w * v.w);
}

// ---------------- pooling + final FC ----------------
__global__ __launch_bounds__(256) void pool_kernel(
    const float* __restrict__ x, const int* __restrict__ batch,
    float* __restrict__ pool, float* __restrict__ counts, int nNodes) {
    const int node = __builtin_amdgcn_readfirstlane(
        (int)((blockIdx.x * blockDim.x + threadIdx.x) >> 5));
    const int lane = threadIdx.x & 31;
    if (node >= nNodes) return;
    const int grp = batch[node];
    if (lane == 0) atomicAdd(counts + grp, 1.0f);
    const float4 v = *(const float4*)(x + (size_t)node * 128 + lane * 4);
    float* dst = pool + (size_t)grp * 128 + lane * 4;
    atomicAdd(dst + 0, v.x);
    atomicAdd(dst + 1, v.y);
    atomicAdd(dst + 2, v.z);
    atomicAdd(dst + 3, v.w);
}

__global__ void final_fc(const float* __restrict__ pool, const float* __restrict__ counts,
                         const float* __restrict__ Wfc, const float* __restrict__ bfc,
                         float* __restrict__ out) {
    const int t = threadIdx.x;            // 128 threads: (g, o)
    if (t >= NG * 2) return;
    const int g = t >> 1, o = t & 1;
    const float inv = 1.0f / fmaxf(counts[g], 1.0f);
    float s = 0.0f;
    for (int c = 0; c < 128; ++c) s += pool[g * 128 + c] * inv * Wfc[o * 128 + c];
    out[g * 2 + o] = s + bfc[o];
}

// ---------------- launch ----------------
extern "C" void kernel_launch(void* const* d_in, const int* in_sizes, int n_in,
                              void* d_out, int out_size, void* d_ws, size_t ws_size,
                              hipStream_t stream) {
    const float* x   = (const float*)d_in[0];
    const int*   ei  = (const int*)d_in[1];      // [2, NE]: row = ei[0:NE], col = ei[NE:2NE]
    const int*   bat = (const int*)d_in[2];
    const float* W1  = (const float*)d_in[3];  const float* b1 = (const float*)d_in[4];
    const float* W2  = (const float*)d_in[5];  const float* b2 = (const float*)d_in[6];
    const float* W3  = (const float*)d_in[7];  const float* b3 = (const float*)d_in[8];
    const float* Wfc = (const float*)d_in[9];  const float* bfc = (const float*)d_in[10];
    const float* g1 = (const float*)d_in[11], *be1 = (const float*)d_in[12],
               *rm1 = (const float*)d_in[13], *rv1 = (const float*)d_in[14];
    const float* g2 = (const float*)d_in[15], *be2 = (const float*)d_in[16],
               *rm2 = (const float*)d_in[17], *rv2 = (const float*)d_in[18];
    const float* g3 = (const float*)d_in[19], *be3 = (const float*)d_in[20],
               *rm3 = (const float*)d_in[21], *rv3 = (const float*)d_in[22];

    char* ws = (char*)d_ws;
    // Region-reused layout (~196 MiB total); all offsets 32B-aligned.
    __bf16* xbf  = (__bf16*)(ws + 0);             // 100000*512*2 = 102,400,000  (dead after L1)
    __bf16* h1   = (__bf16*)(ws + 102400000);     // 51,200,000                  (dead after L2)
    __bf16* h2   = (__bf16*)(ws + 0);             // reuse xbf region            (dead after L3)
    float*  h    = (float*) (ws + 51200000);      // 51,200,000, live through APPNP
    float*  buf0 = (float*) (ws + 102400000);     // reuse h1 region
    float*  buf1 = (float*) (ws + 153600000);     // 51,200,000
    float*  dinv = (float*) (ws + 204800000);     // 400,000
    __bf16* W1b  = (__bf16*)(ws + 205200000);     // 262,144
    __bf16* W2b  = (__bf16*)(ws + 205462144);     // 131,072
    __bf16* W3b  = (__bf16*)(ws + 205593216);     // 65,536
    float*  pool = (float*) (ws + 205658752);     // 64*128*4 = 32,768
    float*  cnts = pool + NG * 128;               // 256

    // --- precision conversions ---
    cvt_f32_bf16<<<(NN * 512 / 4 + 255) / 256, 256, 0, stream>>>(x,  xbf, NN * 512);
    cvt_f32_bf16<<<(256 * 512 / 4 + 255) / 256, 256, 0, stream>>>(W1, W1b, 256 * 512);
    cvt_f32_bf16<<<(256 * 256 / 4 + 255) / 256, 256, 0, stream>>>(W2, W2b, 256 * 256);
    cvt_f32_bf16<<<(128 * 256 / 4 + 255) / 256, 256, 0, stream>>>(W3, W3b, 128 * 256);

    // --- fused MLP (WMMA bf16, f32 accumulate), 32 rows per wave ---
    const int gemmBlocks = (NN / 32 + 7) / 8;     // 8 waves/block, 1 wave per 32-row tile
    gemm_bn_relu<16, true ><<<gemmBlocks, 256, 0, stream>>>(xbf, W1b, b1, g1, be1, rm1, rv1, h1, NN, 512);
    gemm_bn_relu<16, true ><<<gemmBlocks, 256, 0, stream>>>(h1,  W2b, b2, g2, be2, rm2, rv2, h2, NN, 256);
    gemm_bn_relu< 8, false><<<gemmBlocks, 256, 0, stream>>>(h2,  W3b, b3, g3, be3, rm3, rv3, h,  NN, 256);

    // --- degree / normalization (self loops: deg starts at 1) ---
    fill_f32<<<(NN + 255) / 256, 256, 0, stream>>>(dinv, 1.0f, NN);
    deg_edges<<<(NE + 255) / 256, 256, 0, stream>>>(ei + NE, dinv, NE);
    dinv_finalize<<<(NN + 255) / 256, 256, 0, stream>>>(dinv, NN);

    // --- APPNP: 10 ping-pong propagation steps (L2-resident) ---
    const float* cur = h;
    float* nxt = buf0;
    for (int s = 0; s < KSTEPS; ++s) {
        appnp_self<<<(NN * 128 + 255) / 256, 256, 0, stream>>>(cur, h, dinv, nxt, NN * 128);
        appnp_edges<<<(NE * 32 + 255) / 256, 256, 0, stream>>>(ei, ei + NE, dinv, cur, nxt, NE);
        cur = nxt;
        nxt = (nxt == buf0) ? buf1 : buf0;
    }

    // --- mean pooling + final linear ---
    hipMemsetAsync(pool, 0, (NG * 128 + NG) * sizeof(float), stream);
    pool_kernel<<<(NN * 32 + 255) / 256, 256, 0, stream>>>(cur, bat, pool, cnts, NN);
    final_fc<<<1, 128, 0, stream>>>(pool, cnts, Wfc, bfc, (float*)d_out);
}
